// MaskMamba_22892175688210
// MI455X (gfx1250) — matrix operand order repaired and checked by reading, not compiled, not run on hardware
//
#include <hip/hip_runtime.h>
#include <math.h>

// ---------------------------------------------------------------------------
// Types for CDNA5 WMMA (wave32): D(16x16 f32) = A(16x32 bf16) x B(32x16 bf16) + C
// ---------------------------------------------------------------------------
typedef __bf16 bf16;
typedef __bf16 v16bf __attribute__((ext_vector_type(16)));
typedef float  v8f   __attribute__((ext_vector_type(8)));

union V16 { v16bf v; bf16 e[16]; uint4 q[2]; };

// Model constants
#define CB    16
#define CG    256
#define NPT   32
#define GVIS  128
#define CD    384
#define DEPTH 12
#define DI    768
#define DST   16
#define DCV   4
#define DTR   24
#define ENC   384
#define KNN   16
#define TOK   (CB * GVIS)          // 2048 sequence tokens
#define BG    (CB * CG)            // 4096 groups

// ---------------------------------------------------------------------------
// Fragment loads (all unguarded 128-bit loads; K padded to multiple of 32,
// weight buffers padded to Npad rows so every row read is in-bounds).
//  A layout (16x32 bf16): lane L -> row m = L&15; element e -> k = (e&7) + 16*(e>>3) + 8*hi
//  B layout (32x16 bf16): lane L -> col n = L&15; element e -> k = (e&7) + 16*(e>>3) + 8*hi
//  W stored (Npad x K) row-major so B[k][n] = W[n][k] -> contiguous per-lane loads.
// ---------------------------------------------------------------------------
__device__ inline v16bf load_frag_a(const bf16* Arow, int kb, int hi) {
  V16 a;
  a.q[0] = *(const uint4*)(Arow + kb + (hi << 3));
  a.q[1] = *(const uint4*)(Arow + kb + 16 + (hi << 3));
  return a.v;
}
__device__ inline v16bf load_frag_b(const bf16* Wrow, int kb, int hi) {
  V16 b;
  b.q[0] = *(const uint4*)(Wrow + kb + (hi << 3));
  b.q[1] = *(const uint4*)(Wrow + kb + 16 + (hi << 3));
  return b.v;
}
__device__ inline v8f wmma_bf16(v16bf a, v16bf b, v8f acc) {
  return __builtin_amdgcn_wmma_f32_16x16x32_bf16(false, a, false, b,
                                                 (short)0, acc, false, false);
}

__device__ inline float mm_act(float v, int act) {
  if (act == 1) return fmaxf(v, 0.0f);
  if (act == 2) return 0.5f * v * (1.0f + erff(v * 0.70710678118654752f)); // exact gelu
  if (act == 3) return (v > 20.0f) ? v : log1pf(expf(v));                  // softplus
  return v;
}

// ---------------------------------------------------------------------------
// Generic WMMA GEMM:  out = act(A(MxK) @ W(NxK)^T * scale + bias) + resid
// Block: 256 threads = 8 waves; each wave owns a 32x32 C block (2 m-tiles x
// 2 n-tiles; A and B fragments each reused twice -> 2 b128 loads per WMMA).
// Grid: (ceil(N/256), M/32). M multiple of 32; K multiple of 32; W has
// Npad >= roundup(N,32) valid (zero-padded) rows.
// ---------------------------------------------------------------------------
__global__ __launch_bounds__(256)
void mm_gemm_bf16(const bf16* __restrict__ A, const bf16* __restrict__ W,
                  const float* __restrict__ scale, const float* __restrict__ bias,
                  const float* __restrict__ resid,
                  float* __restrict__ outF, bf16* __restrict__ outB,
                  int M, int N, int K, int act) {
  int wave = threadIdx.x >> 5;
  int lane = threadIdx.x & 31;
  int hi   = lane >> 4;
  int ln15 = lane & 15;

  int n0 = (blockIdx.x * 8 + wave) * 32;
  if (n0 >= N) return;
  int m0 = blockIdx.y * 32;

  const bf16* A0 = A + (size_t)(m0 + ln15) * K;
  const bf16* A1 = A + (size_t)(m0 + 16 + ln15) * K;
  const bf16* W0 = W + (size_t)(n0 + ln15) * K;
  const bf16* W1 = W + (size_t)(n0 + 16 + ln15) * K;

  v8f acc[2][2] = {{{}, {}}, {{}, {}}};
  for (int kb = 0; kb < K; kb += 32) {
    __builtin_prefetch((const void*)(W0 + kb + 512), 0, 3);   // global_prefetch_b8 (near)
    __builtin_prefetch((const void*)(W1 + kb + 512), 0, 3);
    v16bf a0 = load_frag_a(A0, kb, hi);
    v16bf a1 = load_frag_a(A1, kb, hi);
    v16bf b0 = load_frag_b(W0, kb, hi);
    v16bf b1 = load_frag_b(W1, kb, hi);
    acc[0][0] = wmma_bf16(a0, b0, acc[0][0]);
    acc[0][1] = wmma_bf16(a0, b1, acc[0][1]);
    acc[1][0] = wmma_bf16(a1, b0, acc[1][0]);
    acc[1][1] = wmma_bf16(a1, b1, acc[1][1]);
  }

#pragma unroll
  for (int mt = 0; mt < 2; ++mt) {
#pragma unroll
    for (int nt = 0; nt < 2; ++nt) {
      int n = n0 + nt * 16 + ln15;
      if (n < N) {
#pragma unroll
        for (int r = 0; r < 8; ++r) {
          int m = m0 + mt * 16 + r + (hi << 3);
          float v = acc[mt][nt][r];
          if (scale) v *= scale[n];
          if (bias)  v += bias[n];
          v = mm_act(v, act);
          size_t o = (size_t)m * N + n;
          if (resid) v += resid[o];
          if (outF) outF[o] = v;
          if (outB) outB[o] = (bf16)v;
        }
      }
    }
  }
}

// ---------------------------------------------------------------------------
// Fused PointNet encoder. One workgroup (256 thr = 8 waves) per (b,g) group.
// Dynamic LDS: bufA(32x512 bf16) | bufB(32x512 bf16) | red(512 f32) = 67584 B
// A fragments come from LDS via ds_load_b128; B fragments stream bf16 weights.
// ---------------------------------------------------------------------------
__global__ __launch_bounds__(256)
void mm_encoder(const float* __restrict__ nb, const float* __restrict__ w1,
                const float* __restrict__ s1, const float* __restrict__ t1,
                const bf16* __restrict__ w2b, const float* __restrict__ b2v,
                const bf16* __restrict__ w3b, const float* __restrict__ s2,
                const float* __restrict__ t2, const bf16* __restrict__ w4b,
                const float* __restrict__ b4v, float* __restrict__ tokens) {
  extern __shared__ char smem[];
  bf16*  bufA = (bf16*)smem;                 // 32x512
  bf16*  bufB = (bf16*)(smem + 32768);       // 32x512
  float* red  = (float*)(smem + 65536);      // 512

  int bg  = blockIdx.x;
  int tid = threadIdx.x;
  int wave = tid >> 5, lane = tid & 31, hi = lane >> 4, ln15 = lane & 15;
  const float* x = nb + (size_t)bg * NPT * 3;

  // ---- layer1: (32x3)@(128x3)^T, bn+relu folded into s1/t1 -> bufA (32x128)
  for (int idx = tid; idx < 32 * 128; idx += 256) {
    int p = idx >> 7, c = idx & 127;
    float f = w1[c * 3 + 0] * x[p * 3 + 0] + w1[c * 3 + 1] * x[p * 3 + 1] +
              w1[c * 3 + 2] * x[p * 3 + 2];
    f = fmaxf(f * s1[c] + t1[c], 0.0f);
    bufA[p * 128 + c] = (bf16)f;
  }
  __syncthreads();

  // ---- layer2: (32x128)@(256x128)^T + b2 -> bufB cols [256..512) (concat tail)
  for (int job = wave; job < 2 * 16; job += 8) {
    int m0 = (job & 1) * 16, n0 = (job >> 1) * 16;
    const bf16* Arow = bufA + (m0 + ln15) * 128;
    const bf16* Wrow = w2b + (size_t)(n0 + ln15) * 128;
    v8f acc = {};
    for (int kb = 0; kb < 128; kb += 32)
      acc = wmma_bf16(load_frag_a(Arow, kb, hi), load_frag_b(Wrow, kb, hi), acc);
#pragma unroll
    for (int r = 0; r < 8; ++r) {
      int m = m0 + r + (hi << 3), n = n0 + ln15;
      bufB[m * 512 + 256 + n] = (bf16)(acc[r] + b2v[n]);
    }
  }
  __syncthreads();

  // ---- global-max over points -> broadcast to concat head (cols [0..256))
  for (int c = tid; c < 256; c += 256) {
    float mx = -3.4e38f;
    for (int p = 0; p < 32; ++p) mx = fmaxf(mx, (float)bufB[p * 512 + 256 + c]);
    red[c] = mx;
  }
  __syncthreads();
  for (int idx = tid; idx < 32 * 256; idx += 256) {
    int p = idx >> 8, c = idx & 255;
    bufB[p * 512 + c] = (bf16)red[c];
  }
  __syncthreads();

  // ---- layer3: (32x512)@(512x512)^T, bn+relu via s2/t2 -> bufA (32x512)
  for (int job = wave; job < 2 * 32; job += 8) {
    int m0 = (job & 1) * 16, n0 = (job >> 1) * 16;
    const bf16* Arow = bufB + (m0 + ln15) * 512;
    const bf16* Wrow = w3b + (size_t)(n0 + ln15) * 512;
    v8f acc = {};
    for (int kb = 0; kb < 512; kb += 32)
      acc = wmma_bf16(load_frag_a(Arow, kb, hi), load_frag_b(Wrow, kb, hi), acc);
#pragma unroll
    for (int r = 0; r < 8; ++r) {
      int m = m0 + r + (hi << 3), n = n0 + ln15;
      bufA[m * 512 + n] = (bf16)fmaxf(acc[r] * s2[n] + t2[n], 0.0f);
    }
  }
  __syncthreads();

  // ---- layer4: (32x512)@(384x512)^T + b4 -> bufB (32x384)
  for (int job = wave; job < 2 * 24; job += 8) {
    int m0 = (job & 1) * 16, n0 = (job >> 1) * 16;
    const bf16* Arow = bufA + (m0 + ln15) * 512;
    const bf16* Wrow = w4b + (size_t)(n0 + ln15) * 512;
    v8f acc = {};
    for (int kb = 0; kb < 512; kb += 32)
      acc = wmma_bf16(load_frag_a(Arow, kb, hi), load_frag_b(Wrow, kb, hi), acc);
#pragma unroll
    for (int r = 0; r < 8; ++r) {
      int m = m0 + r + (hi << 3), n = n0 + ln15;
      bufB[m * 384 + n] = (bf16)(acc[r] + b4v[n]);
    }
  }
  __syncthreads();

  // ---- final max over points -> token
  for (int c = tid; c < ENC; c += 256) {
    float mx = -3.4e38f;
    for (int p = 0; p < 32; ++p) mx = fmaxf(mx, (float)bufB[p * 384 + c]);
    tokens[(size_t)bg * ENC + c] = mx;
  }
}

// ---------------------------------------------------------------------------
// Small utility kernels
// ---------------------------------------------------------------------------
__global__ void mm_cvt_bf16(const float* __restrict__ s, bf16* __restrict__ d, int n) {
  int i = blockIdx.x * blockDim.x + threadIdx.x;
  if (i < n) d[i] = (bf16)s[i];
}

// f32 (N x K) -> bf16 (Npad x Kpad) with zero padding
__global__ void mm_cvt_pad(const float* __restrict__ s, bf16* __restrict__ d,
                           int N, int K, int Npad, int Kpad) {
  int i = blockIdx.x * blockDim.x + threadIdx.x;
  if (i >= Npad * Kpad) return;
  int k = i % Kpad, n = i / Kpad;
  d[i] = (n < N && k < K) ? (bf16)s[(size_t)n * K + k] : (bf16)0.0f;
}

// Fold batchnorm + linear bias into per-channel scale/shift
__global__ void mm_bn_fuse(const float* b1, const float* g1, const float* bb1,
                           const float* m1, const float* v1,
                           const float* b3, const float* g2, const float* bb2,
                           const float* m2, const float* v2,
                           float* s1, float* t1, float* s2, float* t2) {
  int i = blockIdx.x * blockDim.x + threadIdx.x;
  if (i < 128) {
    float s = g1[i] * rsqrtf(v1[i] + 1e-5f);
    s1[i] = s; t1[i] = (b1[i] - m1[i]) * s + bb1[i];
  } else if (i < 128 + 512) {
    int j = i - 128;
    float s = g2[j] * rsqrtf(v2[j] + 1e-5f);
    s2[j] = s; t2[j] = (b3[j] - m2[j]) * s + bb2[j];
  }
}

// kNN (K=16 smallest squared distances, stable ties like jax top_k)
__global__ void mm_knn(const float* __restrict__ center, int* __restrict__ knn) {
  int g = blockIdx.x * blockDim.x + threadIdx.x;
  if (g >= BG) return;
  int b = g >> 8, gi = g & 255;
  const float* cb = center + (size_t)b * CG * 3;
  float cx = cb[gi * 3], cy = cb[gi * 3 + 1], cz = cb[gi * 3 + 2];
  float bd[KNN]; int bi[KNN];
#pragma unroll
  for (int k = 0; k < KNN; ++k) { bd[k] = 3.4e38f; bi[k] = 0; }
  for (int j = 0; j < CG; ++j) {
    float dx = cb[j * 3] - cx, dy = cb[j * 3 + 1] - cy, dz = cb[j * 3 + 2] - cz;
    float d2 = dx * dx + dy * dy + dz * dz;
    if (d2 < bd[KNN - 1]) {
      int pos = KNN - 1;
      while (pos > 0 && d2 < bd[pos - 1]) { bd[pos] = bd[pos - 1]; bi[pos] = bi[pos - 1]; --pos; }
      bd[pos] = d2; bi[pos] = j;
    }
  }
  for (int k = 0; k < KNN; ++k) knn[g * KNN + k] = bi[k];
}

// pooled = sqrt(sum_k feat[idx_k]^2) -> bf16
__global__ void mm_pool(const float* __restrict__ tokens, const int* __restrict__ knn,
                        bf16* __restrict__ pooled) {
  int idx = blockIdx.x * blockDim.x + threadIdx.x;
  if (idx >= BG * ENC) return;
  int c = idx % ENC, bg = idx / ENC, b = bg >> 8;
  float s = 0.0f;
#pragma unroll
  for (int k = 0; k < KNN; ++k) {
    int j = knn[bg * KNN + k];
    float v = tokens[((size_t)b * CG + j) * ENC + c];
    s += v * v;
  }
  pooled[idx] = (bf16)sqrtf(s);
}

// posh = gelu(c_vis @ pe_w1^T + pe_b1) (K=3, done in VALU) -> bf16
__global__ void mm_pe1(const float* __restrict__ center, const int* __restrict__ vis,
                       const float* __restrict__ w1, const float* __restrict__ b1,
                       bf16* __restrict__ out) {
  int idx = blockIdx.x * blockDim.x + threadIdx.x;
  if (idx >= TOK * 128) return;
  int c = idx & 127, t = idx >> 7, b = t >> 7, i = t & 127;
  int vi = vis[b * GVIS + i];
  const float* cp = center + ((size_t)b * CG + vi) * 3;
  float f = w1[c * 3] * cp[0] + w1[c * 3 + 1] * cp[1] + w1[c * 3 + 2] * cp[2] + b1[c];
  f = 0.5f * f * (1.0f + erff(f * 0.70710678118654752f));
  out[idx] = (bf16)f;
}

// h = tokens[vis] * g1 + b1
__global__ void mm_gather_affine(const float* __restrict__ tokens, const int* __restrict__ vis,
                                 const float* __restrict__ g1, const float* __restrict__ b1,
                                 float* __restrict__ h) {
  int idx = blockIdx.x * blockDim.x + threadIdx.x;
  if (idx >= TOK * CD) return;
  int d = idx % CD, t = idx / CD, b = t >> 7, i = t & 127;
  int vi = vis[b * GVIS + i];
  h[idx] = tokens[((size_t)b * CG + vi) * CD + d] * g1[d] + b1[d];
}

// LayerNorm (thread per token), optional second affine, f32/bf16 outputs
__global__ void mm_ln(const float* __restrict__ x, const float* __restrict__ g,
                      const float* __restrict__ b, const float* __restrict__ g2,
                      const float* __restrict__ b2, float* __restrict__ outF,
                      bf16* __restrict__ outB, int T, int D) {
  int t = blockIdx.x * blockDim.x + threadIdx.x;
  if (t >= T) return;
  const float* xr = x + (size_t)t * D;
  float m = 0.0f;
  for (int d = 0; d < D; ++d) m += xr[d];
  m /= (float)D;
  float v = 0.0f;
  for (int d = 0; d < D; ++d) { float c = xr[d] - m; v += c * c; }
  v /= (float)D;
  float rs = rsqrtf(v + 1e-5f);
  for (int d = 0; d < D; ++d) {
    float y = (xr[d] - m) * rs * g[d] + b[d];
    if (g2) y = y * g2[d] + b2[d];
    size_t o = (size_t)t * D + d;
    if (outF) outF[o] = y;
    if (outB) outB[o] = (bf16)y;
  }
}

// Causal depthwise conv (DC=4) + SiLU; input = first DI cols of xz
__global__ void mm_conv_silu(const float* __restrict__ xz, const float* __restrict__ cw,
                             const float* __restrict__ cb, float* __restrict__ xcF,
                             bf16* __restrict__ xcB) {
  int idx = blockIdx.x * blockDim.x + threadIdx.x;
  if (idx >= TOK * DI) return;
  int e = idx % DI, t = idx / DI, l = t & 127;
  float acc = cb[e];
#pragma unroll
  for (int k = 0; k < DCV; ++k) {
    int lm = l + k - (DCV - 1);
    if (lm >= 0) acc += cw[e * DCV + k] * xz[(size_t)(t + k - (DCV - 1)) * (2 * DI) + e];
  }
  float s = acc / (1.0f + expf(-acc));
  xcF[idx] = s;
  xcB[idx] = (bf16)s;
}

// Extract dt_rank slice of dbc -> bf16, zero-padded to K=32
__global__ void mm_split_dt(const float* __restrict__ dbc, bf16* __restrict__ dtp) {
  int idx = blockIdx.x * blockDim.x + threadIdx.x;
  if (idx >= TOK * 32) return;
  int r = idx & 31, t = idx >> 5;
  dtp[idx] = (r < DTR) ? (bf16)dbc[(size_t)t * 56 + r] : (bf16)0.0f;
}

// Selective scan: thread per (b, d), 16 states in registers, B/C via LDS
__global__ __launch_bounds__(256)
void mm_scan(const float* __restrict__ dtf, const float* __restrict__ dbc,
             const float* __restrict__ xz, const float* __restrict__ xcF,
             const float* __restrict__ Alog, const float* __restrict__ Dp,
             bf16* __restrict__ yB) {
  __shared__ float sB[DST], sC[DST];
  int d = blockIdx.x * blockDim.x + threadIdx.x; // 0..767
  int b = blockIdx.y;
  float a[DST], h[DST];
#pragma unroll
  for (int s = 0; s < DST; ++s) { a[s] = -expf(Alog[d * DST + s]); h[s] = 0.0f; }
  float Dd = Dp[d];
  for (int l = 0; l < GVIS; ++l) {
    int t = b * GVIS + l;
    __syncthreads();
    if (threadIdx.x < 32) {
      int s = threadIdx.x & 15;
      if (threadIdx.x < 16) sB[s] = dbc[(size_t)t * 56 + DTR + s];
      else                  sC[s] = dbc[(size_t)t * 56 + DTR + DST + s];
    }
    __syncthreads();
    float dt = dtf[(size_t)t * DI + d];
    float xv = xcF[(size_t)t * DI + d];
    float y = 0.0f;
#pragma unroll
    for (int s = 0; s < DST; ++s) {
      float da = expf(dt * a[s]);
      h[s] = da * h[s] + dt * sB[s] * xv;
      y += h[s] * sC[s];
    }
    y += xv * Dd;
    float z = xz[(size_t)t * (2 * DI) + DI + d];
    y *= z / (1.0f + expf(-z));
    yB[(size_t)t * DI + d] = (bf16)y;
  }
}

// ---------------------------------------------------------------------------
// Host orchestration
// ---------------------------------------------------------------------------
static void launch_gemm(const bf16* A, const bf16* W, const float* scale,
                        const float* bias, const float* resid, float* outF,
                        bf16* outB, int M, int N, int K, int act, hipStream_t s) {
  dim3 grid((N + 255) / 256, M / 32);
  mm_gemm_bf16<<<grid, 256, 0, s>>>(A, W, scale, bias, resid, outF, outB, M, N, K, act);
}

static void cvt(const float* s, bf16* d, int n, hipStream_t st) {
  mm_cvt_bf16<<<(n + 255) / 256, 256, 0, st>>>(s, d, n);
}

extern "C" void kernel_launch(void* const* d_in, const int* in_sizes, int n_in,
                              void* d_out, int out_size, void* d_ws, size_t ws_size,
                              hipStream_t stream) {
  // ---- inputs (setup_inputs dict order) ----
  const float* neighborhood = (const float*)d_in[0];
  const float* center       = (const float*)d_in[1];
  const int*   vis_idx      = (const int*)d_in[2];
  const float* e_w1   = (const float*)d_in[3];
  const float* e_b1   = (const float*)d_in[4];
  const float* e_bn1g = (const float*)d_in[5];
  const float* e_bn1b = (const float*)d_in[6];
  const float* e_bn1m = (const float*)d_in[7];
  const float* e_bn1v = (const float*)d_in[8];
  const float* e_w2   = (const float*)d_in[9];
  const float* e_b2   = (const float*)d_in[10];
  const float* e_w3   = (const float*)d_in[11];
  const float* e_b3   = (const float*)d_in[12];
  const float* e_bn2g = (const float*)d_in[13];
  const float* e_bn2b = (const float*)d_in[14];
  const float* e_bn2m = (const float*)d_in[15];
  const float* e_bn2v = (const float*)d_in[16];
  const float* e_w4   = (const float*)d_in[17];
  const float* e_b4   = (const float*)d_in[18];
  const float* lp_w1  = (const float*)d_in[19];
  const float* lp_b1  = (const float*)d_in[20];
  const float* lp_w2  = (const float*)d_in[21];
  const float* lp_b2  = (const float*)d_in[22];
  const float* pe_w1  = (const float*)d_in[23];
  const float* pe_b1  = (const float*)d_in[24];
  const float* pe_w2  = (const float*)d_in[25];
  const float* pe_b2  = (const float*)d_in[26];
  const float* g1v    = (const float*)d_in[27];
  const float* b1v    = (const float*)d_in[28];
  const float* g2v    = (const float*)d_in[29];
  const float* b2v    = (const float*)d_in[30];
  const float* m_in_w   = (const float*)d_in[31];
  const float* m_conv_w = (const float*)d_in[32];
  const float* m_conv_b = (const float*)d_in[33];
  const float* m_xp_w   = (const float*)d_in[34];
  const float* m_dt_w   = (const float*)d_in[35];
  const float* m_dt_b   = (const float*)d_in[36];
  const float* m_A_log  = (const float*)d_in[37];
  const float* m_D      = (const float*)d_in[38];
  const float* m_out_w  = (const float*)d_in[39];
  const float* ln_g     = (const float*)d_in[40];
  const float* ln_b     = (const float*)d_in[41];
  const float* nf_g     = (const float*)d_in[42];
  const float* nf_b     = (const float*)d_in[43];
  (void)in_sizes; (void)n_in; (void)out_size; (void)ws_size;

  // ---- workspace carving ----
  char* wsb = (char*)d_ws;
  size_t off = 0;
  auto carve = [&](size_t bytes) -> char* {
    char* p = wsb + off;
    off = (off + bytes + 255) & ~(size_t)255;
    return p;
  };
  bf16* w2b   = (bf16*)carve(sizeof(bf16) * 256 * 128);
  bf16* w3b   = (bf16*)carve(sizeof(bf16) * 512 * 512);
  bf16* w4b   = (bf16*)carve(sizeof(bf16) * 384 * 512);
  bf16* lp1b  = (bf16*)carve(sizeof(bf16) * 384 * 384);
  bf16* lp2b  = (bf16*)carve(sizeof(bf16) * 384 * 384);
  bf16* pe2b  = (bf16*)carve(sizeof(bf16) * 384 * 128);
  bf16* inwb  = (bf16*)carve(sizeof(bf16) * DEPTH * 2 * DI * CD);
  bf16* xpwb  = (bf16*)carve(sizeof(bf16) * DEPTH * 64 * DI);   // 56 -> 64 padded rows
  bf16* dtwb  = (bf16*)carve(sizeof(bf16) * DEPTH * DI * 32);   // K 24 -> 32 padded
  bf16* outwb = (bf16*)carve(sizeof(bf16) * DEPTH * CD * DI);
  float* s1 = (float*)carve(sizeof(float) * 128);
  float* t1 = (float*)carve(sizeof(float) * 128);
  float* s2 = (float*)carve(sizeof(float) * 512);
  float* t2 = (float*)carve(sizeof(float) * 512);
  float* tokens  = (float*)carve(sizeof(float) * BG * ENC);
  int*   knn     = (int*)carve(sizeof(int) * BG * KNN);
  bf16*  pooledB = (bf16*)carve(sizeof(bf16) * BG * ENC);
  bf16*  lphB    = (bf16*)carve(sizeof(bf16) * BG * ENC);
  float* hbuf    = (float*)carve(sizeof(float) * TOK * CD);
  bf16*  poshB   = (bf16*)carve(sizeof(bf16) * TOK * 128);
  bf16*  xnB     = (bf16*)carve(sizeof(bf16) * TOK * CD);
  float* xz      = (float*)carve(sizeof(float) * TOK * 2 * DI);
  float* xcF     = (float*)carve(sizeof(float) * TOK * DI);
  bf16*  xcB     = (bf16*)carve(sizeof(bf16) * TOK * DI);
  float* dbc     = (float*)carve(sizeof(float) * TOK * 56);
  bf16*  dtpB    = (bf16*)carve(sizeof(bf16) * TOK * 32);       // K-padded dt slice
  float* dtf     = (float*)carve(sizeof(float) * TOK * DI);
  bf16*  yB      = (bf16*)carve(sizeof(bf16) * TOK * DI);

  // ---- weight prep (f32 -> bf16, padding the odd shapes) + BN folding ----
  cvt(e_w2, w2b, 256 * 128, stream);
  cvt(e_w3, w3b, 512 * 512, stream);
  cvt(e_w4, w4b, 384 * 512, stream);
  cvt(lp_w1, lp1b, 384 * 384, stream);
  cvt(lp_w2, lp2b, 384 * 384, stream);
  cvt(pe_w2, pe2b, 384 * 128, stream);
  cvt(m_in_w, inwb, DEPTH * 2 * DI * CD, stream);
  cvt(m_out_w, outwb, DEPTH * CD * DI, stream);
  for (int i = 0; i < DEPTH; ++i) {
    mm_cvt_pad<<<(64 * DI + 255) / 256, 256, 0, stream>>>(
        m_xp_w + (size_t)i * 56 * DI, xpwb + (size_t)i * 64 * DI, 56, DI, 64, DI);
    mm_cvt_pad<<<(DI * 32 + 255) / 256, 256, 0, stream>>>(
        m_dt_w + (size_t)i * DI * DTR, dtwb + (size_t)i * DI * 32, DI, DTR, DI, 32);
  }
  mm_bn_fuse<<<3, 256, 0, stream>>>(e_b1, e_bn1g, e_bn1b, e_bn1m, e_bn1v,
                                    e_b3, e_bn2g, e_bn2b, e_bn2m, e_bn2v,
                                    s1, t1, s2, t2);

  // ---- encoder (fused, LDS-resident activations, WMMA) ----
  mm_encoder<<<BG, 256, 67584, stream>>>(neighborhood, e_w1, s1, t1, w2b, e_b2,
                                         w3b, s2, t2, w4b, e_b4, tokens);

  // ---- local pool ----
  mm_knn<<<(BG + 255) / 256, 256, 0, stream>>>(center, knn);
  mm_pool<<<(BG * ENC + 255) / 256, 256, 0, stream>>>(tokens, knn, pooledB);
  launch_gemm(pooledB, lp1b, nullptr, lp_b1, nullptr, nullptr, lphB,
              BG, ENC, ENC, /*gelu*/2, stream);
  launch_gemm(lphB, lp2b, nullptr, lp_b2, tokens, tokens, nullptr,
              BG, ENC, ENC, 0, stream);

  // ---- gather visible tokens, positional embedding ----
  mm_gather_affine<<<(TOK * CD + 255) / 256, 256, 0, stream>>>(tokens, vis_idx, g1v, b1v, hbuf);
  mm_pe1<<<(TOK * 128 + 255) / 256, 256, 0, stream>>>(center, vis_idx, pe_w1, pe_b1, poshB);
  launch_gemm(poshB, pe2b, nullptr, pe_b2, hbuf, hbuf, nullptr,
              TOK, CD, 128, 0, stream);

  // ---- 12 Mamba layers ----
  for (int i = 0; i < DEPTH; ++i) {
    mm_ln<<<(TOK + 255) / 256, 256, 0, stream>>>(hbuf, ln_g + i * CD, ln_b + i * CD,
                                                 nullptr, nullptr, nullptr, xnB, TOK, CD);
    launch_gemm(xnB, inwb + (size_t)i * 2 * DI * CD, nullptr, nullptr, nullptr,
                xz, nullptr, TOK, 2 * DI, CD, 0, stream);
    mm_conv_silu<<<(TOK * DI + 255) / 256, 256, 0, stream>>>(
        xz, m_conv_w + (size_t)i * DI * DCV, m_conv_b + (size_t)i * DI, xcF, xcB);
    launch_gemm(xcB, xpwb + (size_t)i * 64 * DI, nullptr, nullptr, nullptr,
                dbc, nullptr, TOK, 56, DI, 0, stream);
    mm_split_dt<<<(TOK * 32 + 255) / 256, 256, 0, stream>>>(dbc, dtpB);
    launch_gemm(dtpB, dtwb + (size_t)i * DI * 32, nullptr, m_dt_b + (size_t)i * DI,
                nullptr, dtf, nullptr, TOK, DI, 32, /*softplus*/3, stream);
    mm_scan<<<dim3(DI / 256, CB), 256, 0, stream>>>(
        dtf, dbc, xz, xcF, m_A_log + (size_t)i * DI * DST, m_D + (size_t)i * DI, yB);
    launch_gemm(yB, outwb + (size_t)i * CD * DI, nullptr, nullptr, hbuf,
                hbuf, nullptr, TOK, CD, DI, 0, stream);
  }

  // ---- final norm + output affine ----
  mm_ln<<<(TOK + 255) / 256, 256, 0, stream>>>(hbuf, nf_g, nf_b, g2v, b2v,
                                               (float*)d_out, nullptr, TOK, CD);
}
